// E3DLSTM_Module_31250182046022
// MI455X (gfx1250) — matrix operand order repaired and checked by reading, not compiled, not run on hardware
//
#include <hip/hip_runtime.h>
#include <hip/hip_bf16.h>
#include <math.h>

// ---------------------------------------------------------------------------
// E3D-LSTM on gfx1250 (MI455X, wave32).
// Conv3d group ops as implicit-GEMM via v_wmma_f32_16x16x32_f16:
//   M = G*32 out-channels, N = B*900 valid positions, K = 32ic*18taps = 576.
// f16 operands are pre-packed (weights once per launch; activations via a
// one-shot im2col in WMMA-native layout) so the GEMM inner loop is pure
// aligned vector loads + WMMA.
// ---------------------------------------------------------------------------

typedef __attribute__((ext_vector_type(16))) _Float16 v16h;
typedef __attribute__((ext_vector_type(8)))  _Float16 v8h;
typedef __attribute__((ext_vector_type(8)))  float    v8f;

#define NB     4
#define NHID   32
#define NPOS   900          // 30*30 valid conv positions
#define NN     3600         // NB * NPOS (GEMM N)
#define NTHW   2048         // 2*32*32
#define NELEM  65536        // 32 * NTHW per batch sample
#define KCONV  576          // 32 ic * 18 taps
#define KTILES 18           // KCONV / 32
#define B16STR (NN * 32)    // halfs per K-tile in packed B (115200)

// nearest-upsample inverse map: element (t,h,w) of the 2x32x32 grid -> 30x30 pos
__device__ __forceinline__ int up_pos(int thw) {
  int hw = thw & 1023;               // t always maps to src t=0
  int hh = hw >> 5;
  int ww = hw & 31;
  return ((hh * 30) >> 5) * 30 + ((ww * 30) >> 5);
}

__device__ __forceinline__ float sigmf(float x) { return 1.0f / (1.0f + expf(-x)); }

__global__ void zero_kernel(float* __restrict__ p, int n) {
  int i = blockIdx.x * blockDim.x + threadIdx.x;
  if (i < n) p[i] = 0.0f;
}

// fp32 -> f16 convert (weights; layout preserved)
__global__ void cvt_kernel(const float* __restrict__ src, _Float16* __restrict__ dst, int n) {
  int i = blockIdx.x * blockDim.x + threadIdx.x;
  if (i < n) dst[i] = (_Float16)src[i];
}

// im2col into WMMA-native f16 B layout: B16[k>>5][n][k&31]
// src: [B][32][2][32][32] fp32;  n = b*900 + oh*30 + ow;  k = ic*18 + kt*9 + rh*3 + kw
__global__ void im2col_kernel(const float* __restrict__ src, _Float16* __restrict__ B16) {
  int e = blockIdx.x * 256 + threadIdx.x;      // < NN * KCONV = 2,073,600
  int n = e / KCONV;
  int k = e - n * KCONV;
  int b = n / NPOS;
  int pos = n - b * NPOS;
  int oh = pos / 30;
  int ow = pos - oh * 30;
  int ic = k / 18;
  int rk = k - ic * 18;
  int kt = rk / 9;
  int r9 = rk - kt * 9;
  int rh = r9 / 3;
  int kw = r9 - rh * 3;
  float v = src[((size_t)(b * 32 + ic) * 2 + kt) * 1024 + (oh + rh) * 32 + (ow + kw)];
  B16[(size_t)(k >> 5) * B16STR + (size_t)n * 32 + (k & 31)] = (_Float16)v;
}

// pack [c;m] transposed to f16: cmT[n][64], n = b*2048 + pos
__global__ void cm_pack_kernel(const float* __restrict__ cnew,
                               const float* __restrict__ mnew,
                               _Float16* __restrict__ cmT) {
  int e = blockIdx.x * 256 + threadIdx.x;      // < NB*NTHW*64 = 524288
  int n = e >> 6;
  int cc = e & 63;
  int b = n >> 11;
  int pos = n & (NTHW - 1);
  float v = (cc < 32) ? cnew[((size_t)b * 32 + cc) * NTHW + pos]
                      : mnew[((size_t)b * 32 + (cc - 32)) * NTHW + pos];
  cmT[e] = (_Float16)v;
}

__device__ __forceinline__ void block_reduce2(float& a, float& b, float* sh) {
  int tid = threadIdx.x;
  sh[tid] = a; sh[256 + tid] = b;
  __syncthreads();
  for (int st = 128; st > 0; st >>= 1) {
    if (tid < st) { sh[tid] += sh[tid + st]; sh[256 + tid] += sh[256 + tid + st]; }
    __syncthreads();
  }
  a = sh[0]; b = sh[256];
}

// ---------------------------------------------------------------------------
// GEMM conv: A16 [GO][576] f16 (row-major), B16 packed (see im2col), dst [B][GO][900]
// grid (GO/16, 225); block 32 (one wave per 16x16 tile)
// ---------------------------------------------------------------------------
__global__ void conv_wmma_kernel(const _Float16* __restrict__ A16,
                                 const _Float16* __restrict__ B16,
                                 const float* __restrict__ bias,
                                 float* __restrict__ dst, int GO) {
  const int lane = threadIdx.x;
  const int lp = lane & 15;
  const int hb = lane >> 4;
  const int m0 = blockIdx.x * 16;
  const int n  = blockIdx.y * 16 + lp;
  const int bi = n / 900;
  const int posn = n - bi * 900;
  const _Float16* arow = A16 + (size_t)(m0 + lp) * KCONV;
  const _Float16* bcol = B16 + (size_t)n * 32 + hb * 16;

  v8f acc = {};
  for (int t = 0; t < KTILES; ++t) {
    const int kk = t * 32;
    v8h a0 = *(const v8h*)(arow + kk + hb * 16);
    v8h a1 = *(const v8h*)(arow + kk + 8 + hb * 16);
    v16h af;
#pragma unroll
    for (int j = 0; j < 8; ++j) { af[j] = a0[j]; af[8 + j] = a1[j]; }
    v16h bf = *(const v16h*)(bcol + (size_t)t * B16STR);
    if (t + 1 < KTILES)
      __builtin_prefetch(bcol + (size_t)(t + 1) * B16STR, 0, 1);
    acc = __builtin_amdgcn_wmma_f32_16x16x32_f16(false, af, false, bf,
                                                 (short)0, acc, false, false);
  }
#pragma unroll
  for (int r = 0; r < 8; ++r) {
    int m = m0 + hb * 8 + r;
    float v = acc[r];
    if (bias) v += bias[m];
    dst[((size_t)bi * GO + m) * NPOS + posn] = v;
  }
}

// ---------------------------------------------------------------------------
// GroupNorm stats for the 6 early gates: z = xs[baseA+c] + (hs|ms)[baseB+c]
// grid: (B, 6); block 256
// ---------------------------------------------------------------------------
__global__ void gn_stats6_kernel(const float* __restrict__ xsb,
                                 const float* __restrict__ hsb,
                                 const float* __restrict__ msb,
                                 float* __restrict__ mean6,
                                 float* __restrict__ rstd6) {
  __shared__ float sh[512];
  const int b = blockIdx.x, g = blockIdx.y;
  const int baseA = g * 32;                       // xr,xi,xg,xip,xgp,xfp
  const float* pb = (g < 3) ? hsb : msb;
  const int strideB = (g < 3) ? 128 : 96;
  const int baseB = ((g < 3) ? g : (g - 3)) * 32; // hr,hi,hg | mi,mg,mf
  float s1 = 0.f, s2 = 0.f;
  for (int i = threadIdx.x; i < NELEM; i += 256) {
    int c = i >> 11;
    int cp = up_pos(i & 2047);
    float v = xsb[((size_t)b * 224 + baseA + c) * NPOS + cp]
            + pb [((size_t)b * strideB + baseB + c) * NPOS + cp];
    s1 += v; s2 += v * v;
  }
  block_reduce2(s1, s2, sh);
  if (threadIdx.x == 0) {
    float mu = s1 * (1.0f / NELEM);
    float var = s2 * (1.0f / NELEM) - mu * mu;
    mean6[g * NB + b] = mu;
    rstd6[g * NB + b] = rsqrtf(var + 1e-5f);
  }
}

// ---------------------------------------------------------------------------
// Gates pass 1: r, i*g, m_new = ip*gp + fp*m
// ---------------------------------------------------------------------------
__global__ void gates1_kernel(const float* __restrict__ xsb,
                              const float* __restrict__ hsb,
                              const float* __restrict__ msb,
                              const float* __restrict__ m_in,
                              const float* __restrict__ mean6,
                              const float* __restrict__ rstd6,
                              const float* __restrict__ gamma,
                              const float* __restrict__ beta,
                              float* __restrict__ rbuf,
                              float* __restrict__ igbuf,
                              float* __restrict__ m_out) {
  int e = blockIdx.x * 256 + threadIdx.x;   // < B*NELEM
  int b = e >> 16;
  int i = e & (NELEM - 1);
  int c = i >> 11;
  int cp = up_pos(i & 2047);
  size_t xb = (size_t)b * 224 * NPOS;
  size_t hbb = (size_t)b * 128 * NPOS;
  size_t mb = (size_t)b * 96 * NPOS;
  float ga = gamma[c], be = beta[c];

  float xr  = xsb[xb + (size_t)(0   + c) * NPOS + cp];
  float xi  = xsb[xb + (size_t)(32  + c) * NPOS + cp];
  float xg  = xsb[xb + (size_t)(64  + c) * NPOS + cp];
  float xip = xsb[xb + (size_t)(96  + c) * NPOS + cp];
  float xgp = xsb[xb + (size_t)(128 + c) * NPOS + cp];
  float xfp = xsb[xb + (size_t)(160 + c) * NPOS + cp];
  float hr  = hsb[hbb + (size_t)(0  + c) * NPOS + cp];
  float hi  = hsb[hbb + (size_t)(32 + c) * NPOS + cp];
  float hg  = hsb[hbb + (size_t)(64 + c) * NPOS + cp];
  float mi  = msb[mb + (size_t)(0  + c) * NPOS + cp];
  float mg  = msb[mb + (size_t)(32 + c) * NPOS + cp];
  float mf  = msb[mb + (size_t)(64 + c) * NPOS + cp];

  float r  = sigmf((xr  + hr - mean6[0 * NB + b]) * rstd6[0 * NB + b] * ga + be);
  float i_ = sigmf((xi  + hi - mean6[1 * NB + b]) * rstd6[1 * NB + b] * ga + be);
  float gg = tanhf((xg  + hg - mean6[2 * NB + b]) * rstd6[2 * NB + b] * ga + be);
  float ip = sigmf((xip + mi - mean6[3 * NB + b]) * rstd6[3 * NB + b] * ga + be);
  float gp = tanhf((xgp + mg - mean6[4 * NB + b]) * rstd6[4 * NB + b] * ga + be);
  float fp = sigmf((xfp + mf - mean6[5 * NB + b]) * rstd6[5 * NB + b] * ga + be);

  rbuf[e]  = r;
  igbuf[e] = i_ * gg;
  m_out[e] = ip * gp + fp * m_in[e];
}

// scores[b][l] = <r, c_hist[logical l]> / sqrt(2048);  grid (B, TAU)
__global__ void att_scores_kernel(const float* __restrict__ rbuf,
                                  const float* __restrict__ chl, int s,
                                  float* __restrict__ scores) {
  __shared__ float sh[512];
  const int b = blockIdx.x, l = blockIdx.y;
  const float* ch = chl + (size_t)((s + l) & 3) * (NB * NELEM) + (size_t)b * NELEM;
  const float* rp = rbuf + (size_t)b * NELEM;
  float s1 = 0.f, s2 = 0.f;
  for (int i = threadIdx.x; i < NELEM; i += 256) s1 += rp[i] * ch[i];
  block_reduce2(s1, s2, sh);
  if (threadIdx.x == 0) scores[b * 4 + l] = s1 * 0.022097086912079608f; // 1/sqrt(2048)
}

// softmax over TAU + stats of (c_last + recall); stores combined weights
__global__ void att_stats_kernel(const float* __restrict__ chl,
                                 const float* __restrict__ scores, int s,
                                 float* __restrict__ attnw,
                                 float* __restrict__ mean4,
                                 float* __restrict__ rstd4) {
  __shared__ float sh[512];
  const int b = blockIdx.x;
  float sc0 = scores[b * 4 + 0], sc1 = scores[b * 4 + 1];
  float sc2 = scores[b * 4 + 2], sc3 = scores[b * 4 + 3];
  float mx = fmaxf(fmaxf(sc0, sc1), fmaxf(sc2, sc3));
  float e0 = expf(sc0 - mx), e1 = expf(sc1 - mx), e2 = expf(sc2 - mx), e3 = expf(sc3 - mx);
  float inv = 1.0f / (e0 + e1 + e2 + e3);
  // combined weights: softmax attn + 1 on logical-last (c_hist[-1])
  float w0 = e0 * inv, w1 = e1 * inv, w2 = e2 * inv, w3 = e3 * inv + 1.0f;
  const float* p0 = chl + (size_t)((s + 0) & 3) * (NB * NELEM) + (size_t)b * NELEM;
  const float* p1 = chl + (size_t)((s + 1) & 3) * (NB * NELEM) + (size_t)b * NELEM;
  const float* p2 = chl + (size_t)((s + 2) & 3) * (NB * NELEM) + (size_t)b * NELEM;
  const float* p3 = chl + (size_t)((s + 3) & 3) * (NB * NELEM) + (size_t)b * NELEM;
  float s1 = 0.f, s2 = 0.f;
  for (int i = threadIdx.x; i < NELEM; i += 256) {
    float v = w0 * p0[i] + w1 * p1[i] + w2 * p2[i] + w3 * p3[i];
    s1 += v; s2 += v * v;
  }
  block_reduce2(s1, s2, sh);
  if (threadIdx.x == 0) {
    float mu = s1 * (1.0f / NELEM);
    float var = s2 * (1.0f / NELEM) - mu * mu;
    mean4[b] = mu;
    rstd4[b] = rsqrtf(var + 1e-5f);
    attnw[b * 4 + 0] = w0; attnw[b * 4 + 1] = w1;
    attnw[b * 4 + 2] = w2; attnw[b * 4 + 3] = w3;
  }
}

// c_new = i*g + gn(c_last + recall); written into ring slot s%4
__global__ void c_update_kernel(float* __restrict__ chl,
                                const float* __restrict__ attnw,
                                const float* __restrict__ igbuf,
                                const float* __restrict__ mean4,
                                const float* __restrict__ rstd4,
                                const float* __restrict__ gamma,
                                const float* __restrict__ beta, int s) {
  int e = blockIdx.x * 256 + threadIdx.x;   // < B*NELEM
  int b = e >> 16;
  int i = e & (NELEM - 1);
  int c = i >> 11;
  size_t bo = (size_t)b * NELEM + i;
  const size_t SL = (size_t)NB * NELEM;
  float w0 = attnw[b * 4 + 0], w1 = attnw[b * 4 + 1];
  float w2 = attnw[b * 4 + 2], w3 = attnw[b * 4 + 3];
  float v = w0 * chl[((s + 0) & 3) * SL + bo] + w1 * chl[((s + 1) & 3) * SL + bo]
          + w2 * chl[((s + 2) & 3) * SL + bo] + w3 * chl[((s + 3) & 3) * SL + bo];
  float xn = (v - mean4[b]) * rstd4[b] * gamma[c] + beta[c];
  chl[(s & 3) * SL + bo] = igbuf[e] + xn;
}

// stats of z8 = xo + ho + co + mo; grid (B)
__global__ void gn_stats8_kernel(const float* __restrict__ xsb,
                                 const float* __restrict__ hsb,
                                 const float* __restrict__ cob,
                                 const float* __restrict__ mob,
                                 float* __restrict__ mean8,
                                 float* __restrict__ rstd8) {
  __shared__ float sh[512];
  const int b = blockIdx.x;
  float s1 = 0.f, s2 = 0.f;
  for (int i = threadIdx.x; i < NELEM; i += 256) {
    int c = i >> 11;
    int cp = up_pos(i & 2047);
    float v = xsb[((size_t)b * 224 + 192 + c) * NPOS + cp]
            + hsb[((size_t)b * 128 + 96 + c) * NPOS + cp]
            + cob[((size_t)b * 32 + c) * NPOS + cp]
            + mob[((size_t)b * 32 + c) * NPOS + cp];
    s1 += v; s2 += v * v;
  }
  block_reduce2(s1, s2, sh);
  if (threadIdx.x == 0) {
    float mu = s1 * (1.0f / NELEM);
    float var = s2 * (1.0f / NELEM) - mu * mu;
    mean8[b] = mu;
    rstd8[b] = rsqrtf(var + 1e-5f);
  }
}

// ---------------------------------------------------------------------------
// h_new = sigmoid(gn8(xo+ho+co+mo)) * tanh(W111 · [c; m] + b111)
// GEMM M=32, N=B*2048, K=64 via WMMA with packed f16 operands; fused epilogue.
// grid (2, 512); block 32
// ---------------------------------------------------------------------------
__global__ void out_wmma_kernel(const _Float16* __restrict__ W16,   // [32][64] f16
                                const float* __restrict__ b111l,    // [32]
                                const _Float16* __restrict__ cmT,   // [N][64] f16
                                const float* __restrict__ xsb,
                                const float* __restrict__ hsb,
                                const float* __restrict__ cob,
                                const float* __restrict__ mob,
                                const float* __restrict__ mean8,
                                const float* __restrict__ rstd8,
                                const float* __restrict__ gamma,
                                const float* __restrict__ beta,
                                float* __restrict__ hout,
                                float* __restrict__ dout, int s) {
  const int lane = threadIdx.x;
  const int lp = lane & 15;
  const int hb = lane >> 4;
  const int m0 = blockIdx.x * 16;
  const int n  = blockIdx.y * 16 + lp;
  const int b = n >> 11;
  const int pos = n & (NTHW - 1);
  const int cp = up_pos(pos);
  const _Float16* arow = W16 + (size_t)(m0 + lp) * 64;
  const _Float16* bcol = cmT + (size_t)n * 64 + hb * 16;

  v8f acc = {};
#pragma unroll
  for (int kk = 0; kk < 64; kk += 32) {
    v8h a0 = *(const v8h*)(arow + kk + hb * 16);
    v8h a1 = *(const v8h*)(arow + kk + 8 + hb * 16);
    v16h af;
#pragma unroll
    for (int j = 0; j < 8; ++j) { af[j] = a0[j]; af[8 + j] = a1[j]; }
    v16h bf = *(const v16h*)(bcol + kk);
    acc = __builtin_amdgcn_wmma_f32_16x16x32_f16(false, af, false, bf,
                                                 (short)0, acc, false, false);
  }
  const float mu = mean8[b], rs = rstd8[b];
#pragma unroll
  for (int r = 0; r < 8; ++r) {
    int och = m0 + hb * 8 + r;
    float th = tanhf(acc[r] + b111l[och]);
    float z8 = xsb[((size_t)b * 224 + 192 + och) * NPOS + cp]
             + hsb[((size_t)b * 128 + 96 + och) * NPOS + cp]
             + cob[((size_t)b * 32 + och) * NPOS + cp]
             + mob[((size_t)b * 32 + och) * NPOS + cp];
    float o = sigmf((z8 - mu) * rs * gamma[och] + beta[och]);
    float hv = o * th;
    hout[((size_t)b * 32 + och) * NTHW + pos] = hv;
    if (dout) dout[((size_t)b * 128 + s * 32 + och) * NTHW + pos] = hv;
  }
}

// ---------------------------------------------------------------------------
extern "C" void kernel_launch(void* const* d_in, const int* in_sizes, int n_in,
                              void* d_out, int out_size, void* d_ws, size_t ws_size,
                              hipStream_t stream) {
  (void)in_sizes; (void)n_in; (void)out_size; (void)ws_size;
  const float* x    = (const float*)d_in[0];
  const float* Wx   = (const float*)d_in[1];
  const float* bx   = (const float*)d_in[2];
  const float* Wh4  = (const float*)d_in[3];
  const float* Wm3  = (const float*)d_in[4];
  const float* Wc   = (const float*)d_in[5];
  const float* Wmo  = (const float*)d_in[6];
  const float* W111 = (const float*)d_in[7];
  const float* b111 = (const float*)d_in[8];
  const float* gng  = (const float*)d_in[9];
  const float* gnb  = (const float*)d_in[10];
  float* out = (float*)d_out;

  float* p = (float*)d_ws;
  size_t off = 0;
  auto alloc = [&](size_t nfl) { float* r = p + off; off += nfl; return r; };
  float* chist  = alloc((size_t)2 * 4 * NB * NELEM);  // [L][TAU][B][32][2048]
  float* hstate = alloc((size_t)2 * NB * NELEM);      // [L][B][32][2048]
  float* mbuf   = alloc((size_t)2 * NB * NELEM);      // ping-pong m
  float* xsb    = alloc((size_t)NB * 224 * NPOS);
  float* hsb    = alloc((size_t)NB * 128 * NPOS);
  float* msb    = alloc((size_t)NB * 96 * NPOS);
  float* cob    = alloc((size_t)NB * 32 * NPOS);
  float* mob    = alloc((size_t)NB * 32 * NPOS);
  float* rbuf   = alloc((size_t)NB * NELEM);
  float* igbuf  = alloc((size_t)NB * NELEM);
  float* st     = alloc(256);
  // f16 regions (allocated in float units; counts are halved element counts)
  _Float16* wx16  = (_Float16*)alloc(129024);   // 2*129024 halfs
  _Float16* wh16  = (_Float16*)alloc(73728);    // 2*73728 halfs
  _Float16* wm16  = (_Float16*)alloc(55296);    // 2*55296 halfs
  _Float16* wc16  = (_Float16*)alloc(18432);    // 2*18432 halfs
  _Float16* wmo16 = (_Float16*)alloc(18432);    // 2*18432 halfs
  _Float16* w11116= (_Float16*)alloc(2048);     // 2*2048 halfs
  _Float16* B16   = (_Float16*)alloc((size_t)KTILES * B16STR / 2);  // 2,073,600 halfs
  _Float16* cmT   = (_Float16*)alloc((size_t)NB * NTHW * 64 / 2);   // 524,288 halfs

  float* mean6 = st;        float* rstd6 = st + 32;
  float* scores = st + 64;  float* attnw = st + 80;
  float* mean4 = st + 96;   float* rstd4 = st + 100;
  float* mean8 = st + 104;  float* rstd8 = st + 108;

  const int STATE = NB * NELEM;                        // 262144 floats
  // zero recurrent state (chist + hstate contiguous) — every launch (replay determinism)
  {
    int n = 2 * 4 * STATE + 2 * STATE;
    zero_kernel<<<(n + 255) / 256, 256, 0, stream>>>(chist, n);
  }
  // one-shot weight conversion to f16 (layout preserved)
  cvt_kernel<<<(258048 + 255) / 256, 256, 0, stream>>>(Wx,   wx16,  258048);
  cvt_kernel<<<(147456 + 255) / 256, 256, 0, stream>>>(Wh4,  wh16,  147456);
  cvt_kernel<<<(110592 + 255) / 256, 256, 0, stream>>>(Wm3,  wm16,  110592);
  cvt_kernel<<<(36864  + 255) / 256, 256, 0, stream>>>(Wc,   wc16,  36864);
  cvt_kernel<<<(36864  + 255) / 256, 256, 0, stream>>>(Wmo,  wmo16, 36864);
  cvt_kernel<<<(4096   + 255) / 256, 256, 0, stream>>>(W111, w11116, 4096);

  const int IM2COL_BLKS = NN * KCONV / 256;            // 8100
  for (int s = 0; s < 4; ++s) {
    for (int l = 0; l < 2; ++l) {
      const float* cur = (l == 0) ? (x + (size_t)s * STATE) : hstate; // layer0 h
      float* chl = chist + (size_t)l * 4 * STATE;
      float* hbuf = hstate + (size_t)l * STATE;
      int cell = s * 2 + l;
      float* m_in  = mbuf + (size_t)(cell & 1) * STATE;
      float* m_out = mbuf + (size_t)((cell + 1) & 1) * STATE;
      if (s == 0)
        zero_kernel<<<STATE / 256, 256, 0, stream>>>(m_in, STATE);

      im2col_kernel<<<IM2COL_BLKS, 256, 0, stream>>>(cur, B16);
      conv_wmma_kernel<<<dim3(14, 225), 32, 0, stream>>>(
          wx16 + (size_t)l * 129024, B16, bx + l * 224, xsb, 224);
      im2col_kernel<<<IM2COL_BLKS, 256, 0, stream>>>(hbuf, B16);
      conv_wmma_kernel<<<dim3(8, 225), 32, 0, stream>>>(
          wh16 + (size_t)l * 73728, B16, nullptr, hsb, 128);
      im2col_kernel<<<IM2COL_BLKS, 256, 0, stream>>>(m_in, B16);
      conv_wmma_kernel<<<dim3(6, 225), 32, 0, stream>>>(
          wm16 + (size_t)l * 55296, B16, nullptr, msb, 96);

      gn_stats6_kernel<<<dim3(NB, 6), 256, 0, stream>>>(xsb, hsb, msb, mean6, rstd6);
      gates1_kernel<<<NB * NELEM / 256, 256, 0, stream>>>(
          xsb, hsb, msb, m_in, mean6, rstd6, gng + l * 32, gnb + l * 32,
          rbuf, igbuf, m_out);

      att_scores_kernel<<<dim3(NB, 4), 256, 0, stream>>>(rbuf, chl, s, scores);
      att_stats_kernel<<<NB, 256, 0, stream>>>(chl, scores, s, attnw, mean4, rstd4);
      c_update_kernel<<<NB * NELEM / 256, 256, 0, stream>>>(
          chl, attnw, igbuf, mean4, rstd4, gng + l * 32, gnb + l * 32, s);

      const float* cnew = chl + (size_t)(s & 3) * STATE;
      im2col_kernel<<<IM2COL_BLKS, 256, 0, stream>>>(cnew, B16);
      conv_wmma_kernel<<<dim3(2, 225), 32, 0, stream>>>(
          wc16 + (size_t)l * 18432, B16, nullptr, cob, 32);
      im2col_kernel<<<IM2COL_BLKS, 256, 0, stream>>>(m_out, B16);
      conv_wmma_kernel<<<dim3(2, 225), 32, 0, stream>>>(
          wmo16 + (size_t)l * 18432, B16, nullptr, mob, 32);

      gn_stats8_kernel<<<NB, 256, 0, stream>>>(xsb, hsb, cob, mob, mean8, rstd8);

      cm_pack_kernel<<<NB * NTHW * 64 / 256, 256, 0, stream>>>(cnew, m_out, cmT);
      out_wmma_kernel<<<dim3(2, 512), 32, 0, stream>>>(
          w11116 + (size_t)l * 2048, b111 + l * 32, cmT,
          xsb, hsb, cob, mob, mean8, rstd8, gng + l * 32, gnb + l * 32,
          hbuf, (l == 1) ? out : nullptr, s);
    }
  }
}
